// CredibilityAwareGraphAttention_541165879889
// MI455X (gfx1250) — matrix operand order repaired and checked by reading, not compiled, not run on hardware
//
#include <hip/hip_runtime.h>
#include <hip/hip_bf16.h>
#include <stdint.h>

#define Bsz 2
#define Nn  4096
#define Hh  256
#define NH  4
#define DH  64
// SCALE = sqrt(64) = 8 -> fold 0.125 into Q

typedef __attribute__((ext_vector_type(16))) _Float16 v16h;
typedef __attribute__((ext_vector_type(8)))  float    v8f;
typedef _Float16 half_t;

__device__ __forceinline__ v8f wmma16x16x32(v16h a, v16h b, v8f c) {
  // D = A(16x32) * B(32x16) + C, f32 accumulate
  return __builtin_amdgcn_wmma_f32_16x16x32_f16(false, a, false, b, (short)0, c, false, false);
}

// ---- CDNA5 async global->LDS DMA (ASYNCcnt-tracked), per ISA 08_async_tensor
__device__ __forceinline__ void async_g2l_b128(const void* gsrc, void* ldst) {
  asm volatile("global_load_async_to_lds_b128 %0, %1, off"
               :: "v"((unsigned int)(uintptr_t)ldst), "v"(gsrc)
               : "memory");
}
__device__ __forceinline__ void wait_async_all() {
  asm volatile("s_wait_asynccnt 0" ::: "memory");
}

// A fragment (16x32 f16), row-major src with leading dim ld (halfs).
__device__ __forceinline__ v16h frag_a_rowmajor(const half_t* __restrict__ src, int ld, int lane) {
  v16h a;
  const int m = lane & 15, hs = lane >> 4;
  const half_t* p = src + (size_t)m * ld;
#pragma unroll
  for (int r = 0; r < 8; ++r) {
    const int k = (hs << 4) + (r << 1);
    a[2 * r]     = p[k];
    a[2 * r + 1] = p[k + 1];
  }
  return a;
}

// B fragment (32x16): element (k,n) at src[n*ldn + k] (k contiguous in memory)
__device__ __forceinline__ v16h frag_b_kcontig(const half_t* __restrict__ src, int ldn, int lane) {
  v16h b;
  const int n = lane & 15, hs = lane >> 4;
  const half_t* p = src + (size_t)n * ldn + (hs << 4);
#pragma unroll
  for (int r = 0; r < 8; ++r) { b[2 * r] = p[2 * r]; b[2 * r + 1] = p[2 * r + 1]; }
  return b;
}

// B fragment (32x16): element (k,n) at src[k*ldk + n] (n contiguous, k strided)
__device__ __forceinline__ v16h frag_b_kstride(const half_t* __restrict__ src, int ldk, int lane) {
  v16h b;
  const int n = lane & 15, hs = lane >> 4;
  const half_t* p = src + (size_t)(hs << 4) * ldk + n;
#pragma unroll
  for (int r = 0; r < 8; ++r) {
    b[2 * r]     = p[(size_t)(2 * r) * ldk];
    b[2 * r + 1] = p[(size_t)(2 * r + 1) * ldk];
  }
  return b;
}

// ---------------------------------------------------------------------------
// 1) f32 -> f16 conversion of h and the four weight matrices
__global__ void k_convert(const float* __restrict__ h,
                          const float* __restrict__ wq, const float* __restrict__ wk,
                          const float* __restrict__ wv, const float* __restrict__ wo,
                          half_t* __restrict__ hh,
                          half_t* __restrict__ wqh, half_t* __restrict__ wkh,
                          half_t* __restrict__ wvh, half_t* __restrict__ woh) {
  const int idx = blockIdx.x * blockDim.x + threadIdx.x;
  if (idx < Bsz * Nn * Hh) hh[idx] = (half_t)h[idx];
  if (idx < Hh * Hh) {
    wqh[idx] = (half_t)wq[idx];
    wkh[idx] = (half_t)wk[idx];
    wvh[idx] = (half_t)wv[idx];
    woh[idx] = (half_t)wo[idx];
  }
}

// ---------------------------------------------------------------------------
// 2) QKV projection -> f16 [B][head][N][dh]; Q pre-scaled by 1/8.
__global__ void k_qkv_proj(const half_t* __restrict__ hh,
                           const half_t* __restrict__ wqh, const half_t* __restrict__ wkh,
                           const half_t* __restrict__ wvh,
                           const float* __restrict__ bq, const float* __restrict__ bk,
                           const float* __restrict__ bv,
                           half_t* __restrict__ Qh, half_t* __restrict__ Kh,
                           half_t* __restrict__ Vh) {
  const int lane = threadIdx.x & 31;
  const int wave = threadIdx.x >> 5;
  const int g0 = blockIdx.x << 4;
  const int o0 = (((blockIdx.y << 2) + wave) << 4);
  const int which = blockIdx.z;
  const half_t* W    = (which == 0) ? wqh : (which == 1) ? wkh : wvh;
  const float*  bias = (which == 0) ? bq  : (which == 1) ? bk  : bv;
  half_t* dst        = (which == 0) ? Qh  : (which == 1) ? Kh  : Vh;
  const float scl = (which == 0) ? 0.125f : 1.0f;

  v8f c = {};
#pragma unroll
  for (int k0 = 0; k0 < Hh; k0 += 32) {
    v16h a = frag_a_rowmajor(hh + (size_t)g0 * Hh + k0, Hh, lane);
    v16h b = frag_b_kcontig(W + (size_t)o0 * Hh + k0, Hh, lane);
    c = wmma16x16x32(a, b, c);
  }
  const int col = lane & 15, hs = lane >> 4;
  const float bb = bias[o0 + col];
#pragma unroll
  for (int r = 0; r < 8; ++r) {
    const int row = g0 + r + (hs << 3);
    const int bIdx = row >> 12, n = row & (Nn - 1);
    const int o = o0 + col;
    const int head = o >> 6, d = o & 63;
    dst[((size_t)(bIdx * NH + head) * Nn + n) * DH + d] = (half_t)((c[r] + bb) * scl);
  }
}

// ---------------------------------------------------------------------------
// 3) Softmax stats: per (b, 32-row tile), one wave per head, two 16-row sets.
//    adj tile + tau_j staged in LDS by async DMA, shared across heads.
__global__ void k_stats(const half_t* __restrict__ Qh, const half_t* __restrict__ Kh,
                        const float* __restrict__ tau, const int* __restrict__ adj,
                        const float* __restrict__ cw, const float* __restrict__ cb,
                        float* __restrict__ rmax, float* __restrict__ rsum) {
  __shared__ __align__(16) int   adjTile[32][32];
  __shared__ float tauj[32];

  const int lane = threadIdx.x & 31;
  const int head = threadIdx.x >> 5;
  const int b  = blockIdx.x / (Nn / 32);
  const int i0 = (blockIdx.x % (Nn / 32)) << 5;
  const int col = lane & 15, hs = lane >> 4;
  const size_t hb = (size_t)(b * NH + head) * Nn;

  v16h qa[2][2];
#pragma unroll
  for (int rs = 0; rs < 2; ++rs) {
    qa[rs][0] = frag_a_rowmajor(Qh + (hb + i0 + (rs << 4)) * DH, DH, lane);
    qa[rs][1] = frag_a_rowmajor(Qh + (hb + i0 + (rs << 4)) * DH + 32, DH, lane);
  }

  const float cwv = cw[head], cbv = cb[head];
  float ticw[2][8], m8[2][8], l8[2][8];
#pragma unroll
  for (int rs = 0; rs < 2; ++rs)
#pragma unroll
    for (int r = 0; r < 8; ++r) {
      ticw[rs][r] = tau[b * Nn + i0 + (rs << 4) + r + (hs << 3)] * cwv;
      m8[rs][r] = -3.0e38f;
      l8[rs][r] = 0.0f;
    }

  for (int j0 = 0; j0 < Nn; j0 += 32) {
    // stage adj 32x32 + tau_j via async DMA (2 b128 chunks per thread)
    {
      const int t = threadIdx.x;
#pragma unroll
      for (int cc = 0; cc < 2; ++cc) {
        const int chunk = t + (cc << 7);
        const int arow = chunk >> 3, qd = (chunk & 7) << 2;
        async_g2l_b128(adj + (size_t)(i0 + arow) * Nn + j0 + qd, &adjTile[arow][qd]);
      }
      if (t < 32) tauj[t] = tau[b * Nn + j0 + t];
      wait_async_all();
    }
    __syncthreads();
    if (j0 + 64 < Nn) __builtin_prefetch(Kh + (hb + j0 + 64) * DH, 0, 1);

#pragma unroll
    for (int s = 0; s < 2; ++s) {
      const int jj = j0 + (s << 4);
      const v16h kb0 = frag_b_kcontig(Kh + (hb + jj) * DH, DH, lane);
      const v16h kb1 = frag_b_kcontig(Kh + (hb + jj) * DH + 32, DH, lane);
      const float tj = tauj[(s << 4) + col];
#pragma unroll
      for (int rs = 0; rs < 2; ++rs) {
        v8f c = {};
        c = wmma16x16x32(qa[rs][0], kb0, c);
        c = wmma16x16x32(qa[rs][1], kb1, c);
#pragma unroll
        for (int r = 0; r < 8; ++r) {
          const int lrow = (rs << 4) + r + (hs << 3);
          const int mk = adjTile[lrow][(s << 4) + col];
          const float sv = (mk != 0) ? (c[r] + ticw[rs][r] * tj + cbv) : -1.0e30f;
          const float mo = m8[rs][r];
          const float mn = fmaxf(mo, sv);
          l8[rs][r] = l8[rs][r] * __expf(mo - mn) + __expf(sv - mn);
          m8[rs][r] = mn;
        }
      }
    }
    __syncthreads();  // protect adjTile/tauj before next staging
  }

  // merge (m,l) across the 16 lanes holding one row
#pragma unroll
  for (int off = 1; off < 16; off <<= 1) {
#pragma unroll
    for (int rs = 0; rs < 2; ++rs)
#pragma unroll
      for (int r = 0; r < 8; ++r) {
        const float mo = __shfl_xor(m8[rs][r], off, 16);
        const float lo = __shfl_xor(l8[rs][r], off, 16);
        const float mn = fmaxf(m8[rs][r], mo);
        l8[rs][r] = l8[rs][r] * __expf(m8[rs][r] - mn) + lo * __expf(mo - mn);
        m8[rs][r] = mn;
      }
  }
  if (col == 0) {
#pragma unroll
    for (int rs = 0; rs < 2; ++rs)
#pragma unroll
      for (int r = 0; r < 8; ++r) {
        const size_t idx = hb + i0 + (rs << 4) + r + (hs << 3);
        rmax[idx] = m8[rs][r];
        rsum[idx] = l8[rs][r];
      }
  }
}

// ---------------------------------------------------------------------------
// 4) Emit pass: normalize, write head-mean attention, accumulate P@V.
__global__ void k_attn(const half_t* __restrict__ Qh, const half_t* __restrict__ Kh,
                       const half_t* __restrict__ Vh,
                       const float* __restrict__ tau, const int* __restrict__ adj,
                       const float* __restrict__ cw, const float* __restrict__ cb,
                       const float* __restrict__ rmax, const float* __restrict__ rsum,
                       half_t* __restrict__ ho, float* __restrict__ meanOut) {
  __shared__ __align__(16) half_t pbuf[NH][32][32];
  __shared__ __align__(16) int    adjTile[32][32];
  __shared__ float tauj[32];

  const int lane = threadIdx.x & 31;
  const int head = threadIdx.x >> 5;
  const int b  = blockIdx.x / (Nn / 32);
  const int i0 = (blockIdx.x % (Nn / 32)) << 5;
  const int col = lane & 15, hs = lane >> 4;
  const size_t hb = (size_t)(b * NH + head) * Nn;

  v16h qa[2][2];
#pragma unroll
  for (int rs = 0; rs < 2; ++rs) {
    qa[rs][0] = frag_a_rowmajor(Qh + (hb + i0 + (rs << 4)) * DH, DH, lane);
    qa[rs][1] = frag_a_rowmajor(Qh + (hb + i0 + (rs << 4)) * DH + 32, DH, lane);
  }

  const float cwv = cw[head], cbv = cb[head];
  float ticw[2][8], m8[2][8], il8[2][8];
#pragma unroll
  for (int rs = 0; rs < 2; ++rs)
#pragma unroll
    for (int r = 0; r < 8; ++r) {
      const int row = i0 + (rs << 4) + r + (hs << 3);
      ticw[rs][r] = tau[b * Nn + row] * cwv;
      m8[rs][r] = rmax[hb + row];
      const float l = rsum[hb + row];
      il8[rs][r] = (l > 0.0f) ? (1.0f / l) : 0.0f;
    }

  v8f o[2][4] = {};

  for (int j0 = 0; j0 < Nn; j0 += 32) {
    // stage adj 32x32 + tau_j via async DMA, shared by all heads
    {
      const int t = threadIdx.x;
#pragma unroll
      for (int cc = 0; cc < 2; ++cc) {
        const int chunk = t + (cc << 7);
        const int arow = chunk >> 3, qd = (chunk & 7) << 2;
        async_g2l_b128(adj + (size_t)(i0 + arow) * Nn + j0 + qd, &adjTile[arow][qd]);
      }
      if (t < 32) tauj[t] = tau[b * Nn + j0 + t];
      wait_async_all();
    }
    __syncthreads();
    if (j0 + 64 < Nn) __builtin_prefetch(Kh + (hb + j0 + 64) * DH, 0, 1);

    // scores -> probabilities into pbuf
#pragma unroll
    for (int s = 0; s < 2; ++s) {
      const int jj = j0 + (s << 4);
      const v16h kb0 = frag_b_kcontig(Kh + (hb + jj) * DH, DH, lane);
      const v16h kb1 = frag_b_kcontig(Kh + (hb + jj) * DH + 32, DH, lane);
      const float tj = tauj[(s << 4) + col];
#pragma unroll
      for (int rs = 0; rs < 2; ++rs) {
        v8f c = {};
        c = wmma16x16x32(qa[rs][0], kb0, c);
        c = wmma16x16x32(qa[rs][1], kb1, c);
#pragma unroll
        for (int r = 0; r < 8; ++r) {
          const int lrow = (rs << 4) + r + (hs << 3);
          const int mk = adjTile[lrow][(s << 4) + col];
          const float sc = (mk != 0) ? (c[r] + ticw[rs][r] * tj + cbv) : -1.0e30f;
          const float p = __expf(sc - m8[rs][r]) * il8[rs][r];
          pbuf[head][lrow][(s << 4) + col] = (half_t)p;
        }
      }
    }
    __syncthreads();

    // head-mean attention: 4 threads cover one 32-col row (coalesced b128s)
    {
      const int t = threadIdx.x;
      const int mrow = t >> 2, c0 = (t & 3) << 3;
      float* mp = meanOut + (size_t)(b * Nn + i0 + mrow) * Nn + j0 + c0;
#pragma unroll
      for (int q = 0; q < 8; ++q) {
        const int cj = c0 + q;
        mp[q] = 0.25f * ((float)pbuf[0][mrow][cj] + (float)pbuf[1][mrow][cj] +
                         (float)pbuf[2][mrow][cj] + (float)pbuf[3][mrow][cj]);
      }
    }

    // P @ V: V fragments shared across both row-sets
    const v16h pa0 = frag_a_rowmajor(&pbuf[head][0][0], 32, lane);
    const v16h pa1 = frag_a_rowmajor(&pbuf[head][16][0], 32, lane);
    const half_t* vb = Vh + (hb + j0) * DH;
#pragma unroll
    for (int d = 0; d < 4; ++d) {
      const v16h vf = frag_b_kstride(vb + (d << 4), DH, lane);
      o[0][d] = wmma16x16x32(pa0, vf, o[0][d]);
      o[1][d] = wmma16x16x32(pa1, vf, o[1][d]);
    }
    __syncthreads();  // protect pbuf/adjTile before next iteration
  }

  // store per-head output, concatenated layout [B][N][H] in f16
#pragma unroll
  for (int rs = 0; rs < 2; ++rs)
#pragma unroll
    for (int r = 0; r < 8; ++r) {
      const int row = i0 + (rs << 4) + r + (hs << 3);
      half_t* dst = ho + (size_t)(b * Nn + row) * Hh + head * DH;
      dst[ 0 + col] = (half_t)o[rs][0][r];
      dst[16 + col] = (half_t)o[rs][1][r];
      dst[32 + col] = (half_t)o[rs][2][r];
      dst[48 + col] = (half_t)o[rs][3][r];
    }
}

// ---------------------------------------------------------------------------
// 5) Output projection: h_out = ho @ Wo^T + bo (f32 out)
__global__ void k_outproj(const half_t* __restrict__ ho, const half_t* __restrict__ woh,
                          const float* __restrict__ bo, float* __restrict__ out) {
  const int lane = threadIdx.x & 31;
  const int wave = threadIdx.x >> 5;
  const int g0 = blockIdx.x << 4;
  const int o0 = (((blockIdx.y << 2) + wave) << 4);
  v8f c = {};
#pragma unroll
  for (int k0 = 0; k0 < Hh; k0 += 32) {
    v16h a = frag_a_rowmajor(ho + (size_t)g0 * Hh + k0, Hh, lane);
    v16h b = frag_b_kcontig(woh + (size_t)o0 * Hh + k0, Hh, lane);
    c = wmma16x16x32(a, b, c);
  }
  const int col = lane & 15, hs = lane >> 4;
  const float bb = bo[o0 + col];
#pragma unroll
  for (int r = 0; r < 8; ++r) {
    out[(size_t)(g0 + r + (hs << 3)) * Hh + o0 + col] = c[r] + bb;
  }
}

// ---------------------------------------------------------------------------
extern "C" void kernel_launch(void* const* d_in, const int* in_sizes, int n_in,
                              void* d_out, int out_size, void* d_ws, size_t ws_size,
                              hipStream_t stream) {
  const float* h   = (const float*)d_in[0];
  const float* tau = (const float*)d_in[1];
  const int*   adj = (const int*)d_in[2];
  const float* Wq  = (const float*)d_in[3];
  const float* bq  = (const float*)d_in[4];
  const float* Wk  = (const float*)d_in[5];
  const float* bk  = (const float*)d_in[6];
  const float* Wv  = (const float*)d_in[7];
  const float* bv  = (const float*)d_in[8];
  const float* cw  = (const float*)d_in[9];
  const float* cb  = (const float*)d_in[10];
  const float* Wo  = (const float*)d_in[11];
  const float* bo  = (const float*)d_in[12];
  (void)in_sizes; (void)n_in; (void)out_size; (void)ws_size;

  char* ws = (char*)d_ws;
  size_t off = 0;
  auto carve = [&](size_t bytes) {
    char* p = ws + off;
    off += (bytes + 255) & ~(size_t)255;
    return p;
  };
  half_t* hh   = (half_t*)carve((size_t)Bsz * Nn * Hh * sizeof(half_t));
  half_t* wqh  = (half_t*)carve((size_t)Hh * Hh * sizeof(half_t));
  half_t* wkh  = (half_t*)carve((size_t)Hh * Hh * sizeof(half_t));
  half_t* wvh  = (half_t*)carve((size_t)Hh * Hh * sizeof(half_t));
  half_t* woh  = (half_t*)carve((size_t)Hh * Hh * sizeof(half_t));
  half_t* Qh   = (half_t*)carve((size_t)Bsz * NH * Nn * DH * sizeof(half_t));
  half_t* Kh   = (half_t*)carve((size_t)Bsz * NH * Nn * DH * sizeof(half_t));
  half_t* Vh   = (half_t*)carve((size_t)Bsz * NH * Nn * DH * sizeof(half_t));
  half_t* ho   = (half_t*)carve((size_t)Bsz * Nn * Hh * sizeof(half_t));
  float*  rmax = (float*)carve((size_t)Bsz * NH * Nn * sizeof(float));
  float*  rsum = (float*)carve((size_t)Bsz * NH * Nn * sizeof(float));

  float* outMain = (float*)d_out;
  float* meanOut = outMain + (size_t)Bsz * Nn * Hh;

  const int rowTiles16 = (Bsz * Nn) / 16;  // 512
  const int rowTiles32 = (Bsz * Nn) / 32;  // 256

  k_convert<<<(Bsz * Nn * Hh + 255) / 256, 256, 0, stream>>>(
      h, Wq, Wk, Wv, Wo, hh, wqh, wkh, wvh, woh);

  k_qkv_proj<<<dim3(rowTiles16, Hh / 64, 3), 128, 0, stream>>>(
      hh, wqh, wkh, wvh, bq, bk, bv, Qh, Kh, Vh);

  k_stats<<<dim3(rowTiles32), 128, 0, stream>>>(
      Qh, Kh, tau, adj, cw, cb, rmax, rsum);

  k_attn<<<dim3(rowTiles32), 128, 0, stream>>>(
      Qh, Kh, Vh, tau, adj, cw, cb, rmax, rsum, ho, meanOut);

  k_outproj<<<dim3(rowTiles16, Hh / 64), 128, 0, stream>>>(
      ho, woh, bo, outMain);
}
